// DeformableConv2d_3435973837199
// MI455X (gfx1250) — matrix operand order repaired and checked by reading, not compiled, not run on hardware
//
#include <hip/hip_runtime.h>
#include <math.h>

// ---------------------------------------------------------------------------
// Deformable 1-D conv (W==1) for MI455X / gfx1250 (wave32, WMMA).
//
// Shapes: B=32, C=128, O=128, K=3, L=4096, W=1, PAD=1, DIL=1.
// W==1 collapses the x-direction bilinear to wx = max(0, 1-|off_x|), leaving
// 2 y-corner gathers per (b,k,h,c). Main contraction is a GEMM:
//   out[b,o,h] = sum_{c,k} val[c,k,h]*w_reg[o,c,k]
// M=O=128, N=B*L=131072, Kdim=C*K=384 -> 12.9 GFLOP, gather/L2-bound
// (x = 64MB fits in 192MB L2). WMMA f32_16x16x32_f16, f32 accumulate.
//
// K-dim is reordered to k*128+c so each 32-wide chunk has a uniform tap k:
// no div-by-3, gather metadata lives in registers, and the 4 gathers per
// thread per chunk share one base pointer with 128KB immediate offsets.
// LDS val tile is double-buffered; gathers overlap DS loads + WMMA.
// ---------------------------------------------------------------------------

#define BB   32
#define CC   128
#define OO   128
#define KK   3
#define LL   4096
#define CK   (CC*KK)          // 384
#define NCHUNK (CK/32)        // 12
#define MAXOFF 1024.0f        // max(h,w)/4

typedef __attribute__((ext_vector_type(16))) _Float16 v16h;
typedef __attribute__((ext_vector_type(8)))  _Float16 v8h;
typedef __attribute__((ext_vector_type(8)))  float    v8f;

// ---------------------------------------------------------------------------
// Kernel 0: w_reg (O,C,K) f32 -> f16, reordered flat [o][k*128+c]
// (96KB, L2-resident, shared by all GEMM blocks)
// ---------------------------------------------------------------------------
__global__ void wconv_kernel(const float* __restrict__ w_reg,
                             _Float16* __restrict__ Wh) {
  int i = blockIdx.x * 256 + threadIdx.x;     // i = o*384 + c*3 + k
  if (i < OO * CK) {
    int o = i / CK;
    int r = i - o * CK;
    int c = r / 3;
    int k = r - c * 3;
    Wh[(size_t)o * CK + k * CC + c] = (_Float16)w_reg[i];
  }
}

// ---------------------------------------------------------------------------
// Kernel 1: 9-channel conv (6 offset + 3 mask) + gather metadata.
// Per (b,k,h) store: clamped iy0, and fused weights wf0/wf1 =
//   wy{0,1} * validy{0,1} * wx * mask   (mask = 2*sigmoid(conv_mod)).
// ---------------------------------------------------------------------------
__global__ void offmask_kernel(const float* __restrict__ x,
                               const float* __restrict__ w_off,
                               const float* __restrict__ b_off,
                               const float* __restrict__ w_mod,
                               const float* __restrict__ b_mod,
                               int*   __restrict__ iy0g,
                               float* __restrict__ wf0g,
                               float* __restrict__ wf1g) {
  __shared__ float w9[9 * CK];           // 13.8KB: [j][c*3+q], j<6 off, j>=6 mod
  const int tid = threadIdx.x;
  for (int i = tid; i < 6 * CK; i += 256) w9[i] = w_off[i];
  for (int i = tid; i < 3 * CK; i += 256) w9[6 * CK + i] = w_mod[i];
  __syncthreads();

  const int gid = blockIdx.x * 256 + tid;
  const int b = gid / LL;
  const int h = gid % LL;

  float acc[9];
#pragma unroll
  for (int j = 0; j < 9; j++) acc[j] = 0.0f;

  const float* xb = x + (size_t)b * CC * LL;
  for (int c = 0; c < CC; c++) {
    const float* xc = xb + (size_t)c * LL;
    float xm = (h > 0)      ? xc[h - 1] : 0.0f;   // PAD=1 zero padding
    float x0 = xc[h];
    float xp = (h < LL - 1) ? xc[h + 1] : 0.0f;
#pragma unroll
    for (int j = 0; j < 9; j++) {
      const float* wj = &w9[j * CK + c * 3];
      acc[j] = fmaf(wj[0], xm, acc[j]);
      acc[j] = fmaf(wj[1], x0, acc[j]);
      acc[j] = fmaf(wj[2], xp, acc[j]);
    }
  }

#pragma unroll
  for (int k = 0; k < 3; k++) {
    float oy = acc[2 * k]     + b_off[2 * k];
    float ox = acc[2 * k + 1] + b_off[2 * k + 1];
    oy = fminf(fmaxf(oy, -MAXOFF), MAXOFF);
    ox = fminf(fmaxf(ox, -MAXOFF), MAXOFF);
    float m  = 2.0f / (1.0f + expf(-(acc[6 + k] + b_mod[k])));
    float py = (float)(h + k - 1) + oy;           // base_y + off_y
    float fy = floorf(py);
    int   iy0 = (int)fy;
    float wy1 = py - fy;
    float wy0 = 1.0f - wy1;
    float wx  = fmaxf(0.0f, 1.0f - fabsf(ox));    // W==1 bilinear collapse
    float v0 = (iy0     >= 0 && iy0     < LL) ? 1.0f : 0.0f;
    float v1 = (iy0 + 1 >= 0 && iy0 + 1 < LL) ? 1.0f : 0.0f;
    int iyc = iy0 < 0 ? 0 : (iy0 > LL - 1 ? LL - 1 : iy0);
    size_t idx = ((size_t)b * 3 + k) * LL + h;
    iy0g[idx] = iyc;
    wf0g[idx] = wy0 * v0 * wx * m;
    wf1g[idx] = wy1 * v1 * wx * m;
  }
}

// ---------------------------------------------------------------------------
// Kernel 2: gather + WMMA GEMM, double-buffered, fully unrolled pipeline.
// Block: 256 threads (8 waves), one b, 32 h positions, all 128 o.
// Wave w owns o-tile [16w,16w+16) x both 16-h n-tiles (2 x v8f accumulators).
//   A (16x32 f16) = Wh rows     : lanes 0-15 K{0-7,16-23}, 16-31 K{8-15,24-31}
//   B (32x16 f16) = val columns : lanes 0-15 K{0-7,16-23}, 16-31 K{8-15,24-31}
//   D (16x16 f32): VGPR r: lanes 0-15 M=r,N=lane; lanes 16-31 M=r+8,N=lane-16
// ---------------------------------------------------------------------------
__global__ void __launch_bounds__(256)
deform_gemm_kernel(const float* __restrict__ x,
                   const _Float16* __restrict__ Wh,
                   const int*   __restrict__ iy0g,
                   const float* __restrict__ wf0g,
                   const float* __restrict__ wf1g,
                   float* __restrict__ out) {
  __shared__ __align__(16) _Float16 valLds[2][32][40];  // 2 x 2.5KB, 80B rows

  const int tid   = threadIdx.x;
  const int b     = blockIdx.x / (LL / 32);
  const int htile = (blockIdx.x % (LL / 32)) * 32;

  const int lane    = tid & 31;
  const int wave    = tid >> 5;        // 0..7 -> o tile
  const int ot      = wave * 16;
  const int halfSel = lane >> 4;       // 0: lanes 0-15, 1: lanes 16-31
  const int r16     = lane & 15;
  const int hB      = lane;            // build role: lane -> h (coalesced)
  const int ckB     = wave;            // build role: base ck within chunk

  const float* xb = x + (size_t)b * CC * LL;

  // ---- gather metadata in registers: 3 taps for this thread's h ----
  const float* p0[3];
  const float* p1[3];
  float w0r[3], w1r[3];
#pragma unroll
  for (int k = 0; k < 3; k++) {
    size_t idx = ((size_t)b * 3 + k) * LL + htile + hB;
    int iy  = iy0g[idx];                         // clamped to [0, L-1]
    int iy1 = iy + 1 > LL - 1 ? LL - 1 : iy + 1;
    p0[k] = xb + iy;
    p1[k] = xb + iy1;
    w0r[k] = wf0g[idx];
    w1r[k] = wf1g[idx];
  }

  v8f acc0 = {};
  v8f acc1 = {};

  // ---- prologue: build chunk 0 (tap k=0, c base 0) into buffer 0 ----
  {
    const float* g0p = p0[0] + (size_t)ckB * LL;
    const float* g1p = p1[0] + (size_t)ckB * LL;
#pragma unroll
    for (int p = 0; p < 4; p++) {
      float v = w0r[0] * g0p[(size_t)p * 8 * LL] + w1r[0] * g1p[(size_t)p * 8 * LL];
      valLds[0][hB][ckB + p * 8] = (_Float16)v;
    }
  }
  __syncthreads();

  // ---- steady state: 11 pipelined chunks, fully unrolled ----
#pragma unroll
  for (int it = 0; it < NCHUNK - 1; ++it) {
    const int kk  = it * 32;
    const int cur = it & 1;
    const int kn  = (it + 1) >> 2;         // next chunk's tap (compile-time)
    const int cbn = ((it + 1) & 3) * 32;   // next chunk's c base

    // (1) issue next chunk's gathers early: one clause of 8 global_load_b32
    float g0[4], g1[4];
    {
      const float* g0p = p0[kn] + (size_t)(cbn + ckB) * LL;
      const float* g1p = p1[kn] + (size_t)(cbn + ckB) * LL;
#pragma unroll
      for (int p = 0; p < 4; p++) {
        g0[p] = g0p[(size_t)p * 8 * LL];
        g1[p] = g1p[(size_t)p * 8 * LL];
      }
    }

    // (2) A fragment (W rows for this o-tile) from L2
    const _Float16* Wrow = Wh + (size_t)(ot + r16) * CK + kk + halfSel * 8;
    v8h aLo = *(const v8h*)(Wrow);
    v8h aHi = *(const v8h*)(Wrow + 16);
    v16h a;
#pragma unroll
    for (int i = 0; i < 8; i++) { a[i] = aLo[i]; a[8 + i] = aHi[i]; }

    //     B fragments (val columns) from current LDS buffer, two n-tiles
    const _Float16* Brow0 = &valLds[cur][r16][halfSel * 8];
    const _Float16* Brow1 = &valLds[cur][16 + r16][halfSel * 8];
    v8h b0Lo = *(const v8h*)(Brow0);
    v8h b0Hi = *(const v8h*)(Brow0 + 16);
    v8h b1Lo = *(const v8h*)(Brow1);
    v8h b1Hi = *(const v8h*)(Brow1 + 16);
    v16h b0, b1;
#pragma unroll
    for (int i = 0; i < 8; i++) {
      b0[i] = b0Lo[i]; b0[8 + i] = b0Hi[i];
      b1[i] = b1Lo[i]; b1[8 + i] = b1Hi[i];
    }

    acc0 = __builtin_amdgcn_wmma_f32_16x16x32_f16(false, a, false, b0,
                                                  (short)0, acc0, false, false);
    acc1 = __builtin_amdgcn_wmma_f32_16x16x32_f16(false, a, false, b1,
                                                  (short)0, acc1, false, false);

    // (3) convert + store next chunk into the other buffer
#pragma unroll
    for (int p = 0; p < 4; p++) {
      float v = w0r[kn] * g0[p] + w1r[kn] * g1[p];
      valLds[cur ^ 1][hB][ckB + p * 8] = (_Float16)v;
    }
    __syncthreads();
  }

  // ---- epilogue: last chunk (buffer 1), no build ----
  {
    const int kk  = (NCHUNK - 1) * 32;
    const int cur = (NCHUNK - 1) & 1;
    const _Float16* Wrow = Wh + (size_t)(ot + r16) * CK + kk + halfSel * 8;
    v8h aLo = *(const v8h*)(Wrow);
    v8h aHi = *(const v8h*)(Wrow + 16);
    v16h a;
#pragma unroll
    for (int i = 0; i < 8; i++) { a[i] = aLo[i]; a[8 + i] = aHi[i]; }
    const _Float16* Brow0 = &valLds[cur][r16][halfSel * 8];
    const _Float16* Brow1 = &valLds[cur][16 + r16][halfSel * 8];
    v8h b0Lo = *(const v8h*)(Brow0);
    v8h b0Hi = *(const v8h*)(Brow0 + 16);
    v8h b1Lo = *(const v8h*)(Brow1);
    v8h b1Hi = *(const v8h*)(Brow1 + 16);
    v16h b0, b1;
#pragma unroll
    for (int i = 0; i < 8; i++) {
      b0[i] = b0Lo[i]; b0[8 + i] = b0Hi[i];
      b1[i] = b1Lo[i]; b1[8 + i] = b1Hi[i];
    }
    acc0 = __builtin_amdgcn_wmma_f32_16x16x32_f16(false, a, false, b0,
                                                  (short)0, acc0, false, false);
    acc1 = __builtin_amdgcn_wmma_f32_16x16x32_f16(false, a, false, b1,
                                                  (short)0, acc1, false, false);
  }

  // ---- store: N (=h) striped across lanes -> 64B contiguous segments ----
  float* outb = out + (size_t)b * OO * LL;
  {
    int h0 = htile + r16;
    int h1 = htile + 16 + r16;
#pragma unroll
    for (int r = 0; r < 8; r++) {
      int o = ot + r + halfSel * 8;
      outb[(size_t)o * LL + h0] = acc0[r];
      outb[(size_t)o * LL + h1] = acc1[r];
    }
  }
}

// ---------------------------------------------------------------------------
// Launch. Inputs: x, w_off, b_off, w_mod, b_mod, w_reg (all f32).
// Workspace layout: Wh f16 [128][384] | iy0 int [B*3*L] | wf0 f32 | wf1 f32.
// ---------------------------------------------------------------------------
extern "C" void kernel_launch(void* const* d_in, const int* in_sizes, int n_in,
                              void* d_out, int out_size, void* d_ws, size_t ws_size,
                              hipStream_t stream) {
  const float* x     = (const float*)d_in[0];
  const float* w_off = (const float*)d_in[1];
  const float* b_off = (const float*)d_in[2];
  const float* w_mod = (const float*)d_in[3];
  const float* b_mod = (const float*)d_in[4];
  const float* w_reg = (const float*)d_in[5];
  float* out = (float*)d_out;

  char* ws = (char*)d_ws;
  const size_t whBytes  = (size_t)OO * CK * sizeof(_Float16);     // 98304
  const size_t metaElts = (size_t)BB * 3 * LL;                    // 393216
  _Float16* Wh   = (_Float16*)ws;
  int*      iy0g = (int*)  (ws + whBytes);
  float*    wf0g = (float*)(ws + whBytes + metaElts * 4);
  float*    wf1g = (float*)(ws + whBytes + 2 * metaElts * 4);

  wconv_kernel<<<(OO * CK + 255) / 256, 256, 0, stream>>>(w_reg, Wh);
  offmask_kernel<<<(BB * LL) / 256, 256, 0, stream>>>(
      x, w_off, b_off, w_mod, b_mod, iy0g, wf0g, wf1g);
  deform_gemm_kernel<<<BB * (LL / 32), 256, 0, stream>>>(
      x, Wh, iy0g, wf0g, wf1g, out);
}
